// GinNet_75050258530749
// MI455X (gfx1250) — matrix operand order repaired and checked by reading, not compiled, not run on hardware
//
#include <hip/hip_runtime.h>
#include <hip/hip_bf16.h>

// ---------------------------------------------------------------------------
// GIN network on MI455X (gfx1250): edge scatter-add via f32 global atomics,
// node GEMMs via v_wmma_f32_16x16x32_f16 (wave32), LDS-staged weights.
// ---------------------------------------------------------------------------

#define N_NODES  100000
#define N_EDGES  1600000
#define N_GRAPHS 1000
#define X_S      128
#define HID      64

typedef __attribute__((ext_vector_type(16))) _Float16 v16h;
typedef __attribute__((ext_vector_type(8)))  _Float16 v8h;
typedef __attribute__((ext_vector_type(8)))  float    v8f;

static __device__ __forceinline__ void atomAddF(float* p, float v) {
    __hip_atomic_fetch_add(p, v, __ATOMIC_RELAXED, __HIP_MEMORY_SCOPE_AGENT);
}

static __device__ __forceinline__ v8h cvt8(v8f a) {
    v8h r;
#pragma unroll
    for (int i = 0; i < 8; ++i) r[i] = (_Float16)a[i];
    return r;
}

static __device__ __forceinline__ v16h cat8(v8h lo, v8h hi) {
    v16h r;
#pragma unroll
    for (int i = 0; i < 8; ++i) { r[i] = lo[i]; r[i + 8] = hi[i]; }
    return r;
}

// ---------------------------------------------------------------- zero fill
__global__ void k_zero(float* p, int n) {
    int i = blockIdx.x * blockDim.x + threadIdx.x;
    if (i < n) p[i] = 0.0f;
}

// ------------------------------------------- weight transpose + f32 -> f16
// W  : [K][64] row-major (fin, fout)   ->   Wt : [64][K] (n-major, k contig)
__global__ void k_wcvt(const float* __restrict__ W, _Float16* __restrict__ Wt, int K) {
    int idx = blockIdx.x * blockDim.x + threadIdx.x;
    if (idx >= 64 * K) return;
    int n = idx / K;
    int k = idx - n * K;
    Wt[n * K + k] = (_Float16)W[k * 64 + n];
}

// --------------------------------------------------- edge scatter aggregate
// agg[dst] += h[src], feature group of 4 per blockIdx.y
__global__ void k_edge_agg(const float* __restrict__ h, const int* __restrict__ ei,
                           float* __restrict__ agg, int K) {
    int e = blockIdx.x * blockDim.x + threadIdx.x;
    if (e >= N_EDGES) return;
    int g4 = blockIdx.y * 4;
    int s = ei[e];
    int d = ei[N_EDGES + e];
    const float* hp = h + s * K + g4;
    float*       ap = agg + d * K + g4;
    float m0 = hp[0], m1 = hp[1], m2 = hp[2], m3 = hp[3];
    atomAddF(ap + 0, m0);
    atomAddF(ap + 1, m1);
    atomAddF(ap + 2, m2);
    atomAddF(ap + 3, m3);
}

// ------------------------------------------------------------- WMMA GEMM
// out[m][0..63] = relu( ((1+eps)*h[m][:] + agg[m][:]) @ Wt^T + bias )
// One wave -> 16x64 tile. Weights (f16, transposed) staged in padded LDS.
__global__ void k_gin_gemm(const float* __restrict__ h, const float* __restrict__ agg,
                           const _Float16* __restrict__ Wt, const float* __restrict__ bias,
                           const float* __restrict__ epsp, float* __restrict__ out,
                           int K, int Mtiles) {
    __shared__ _Float16 ldsW[64 * (X_S + 8)];   // max K=128, stride K+8 halves

    const int tid  = threadIdx.x;
    const int lane = tid & 31;
    const int lo16 = lane & 15;
    const int hi   = lane >> 4;           // 0 or 1
    const int ldst = K + 8;               // padded LDS row stride (halves)

    // cooperative load of Wt into LDS (8-half chunks)
    {
        const int kc8 = K >> 3;           // chunks per row
        for (int c = tid; c < 64 * kc8; c += blockDim.x) {
            int n  = c / kc8;
            int kc = (c - n * kc8) << 3;
            *(v8h*)&ldsW[n * ldst + kc] = *(const v8h*)&Wt[n * K + kc];
        }
    }
    __syncthreads();

    const int mtile = blockIdx.x * (blockDim.x >> 5) + (tid >> 5);
    if (mtile >= Mtiles) return;

    const float eps1 = 1.0f + epsp[0];
    const int   row  = mtile * 16 + lo16;

    v8f zero = {};
    v8f acc0 = zero, acc1 = zero, acc2 = zero, acc3 = zero;

    for (int kb = 0; kb < K; kb += 32) {
        // A fragment: lane holds row `row`, halves K = kb+8*hi..+7 and kb+16+8*hi..+7
        int base = row * K + kb + 8 * hi;
        v8f h0 = *(const v8f*)(h   + base);
        v8f h1 = *(const v8f*)(h   + base + 16);
        v8f g0 = *(const v8f*)(agg + base);
        v8f g1 = *(const v8f*)(agg + base + 16);
        v16h A = cat8(cvt8(h0 * eps1 + g0), cvt8(h1 * eps1 + g1));

        // B fragments: lane = column, K-halves kb+8*hi..+7 and kb+16+8*hi..+7 from padded LDS
        const _Float16* wb = &ldsW[lo16 * ldst + kb + 8 * hi];
        {
            const _Float16* wp = wb;                 // nt = 0
            v16h B = cat8(*(const v8h*)wp, *(const v8h*)(wp + 16));
            acc0 = __builtin_amdgcn_wmma_f32_16x16x32_f16(false, A, false, B,
                                                          (short)0, acc0, false, false);
        }
        {
            const _Float16* wp = wb + 16 * ldst;     // nt = 1
            v16h B = cat8(*(const v8h*)wp, *(const v8h*)(wp + 16));
            acc1 = __builtin_amdgcn_wmma_f32_16x16x32_f16(false, A, false, B,
                                                          (short)0, acc1, false, false);
        }
        {
            const _Float16* wp = wb + 32 * ldst;     // nt = 2
            v16h B = cat8(*(const v8h*)wp, *(const v8h*)(wp + 16));
            acc2 = __builtin_amdgcn_wmma_f32_16x16x32_f16(false, A, false, B,
                                                          (short)0, acc2, false, false);
        }
        {
            const _Float16* wp = wb + 48 * ldst;     // nt = 3
            v16h B = cat8(*(const v8h*)wp, *(const v8h*)(wp + 16));
            acc3 = __builtin_amdgcn_wmma_f32_16x16x32_f16(false, A, false, B,
                                                          (short)0, acc3, false, false);
        }
    }

    // epilogue: C/D layout — VGPR r, lanes 0-15: M=r, lanes 16-31: M=r+8; N=lane&15
    v8f accs[4] = {acc0, acc1, acc2, acc3};
#pragma unroll
    for (int nt = 0; nt < 4; ++nt) {
        int   col = nt * 16 + lo16;
        float bv  = bias[col];
#pragma unroll
        for (int r = 0; r < 8; ++r) {
            int   m = mtile * 16 + r + 8 * hi;
            float v = accs[nt][r] + bv;
            out[m * 64 + col] = v > 0.0f ? v : 0.0f;
        }
    }
}

// ------------------------------------------------------------- mean pooling
__global__ void k_pool_sum(const float* __restrict__ h, const int* __restrict__ batch,
                           float* __restrict__ sums) {
    int idx = blockIdx.x * blockDim.x + threadIdx.x;
    if (idx >= N_NODES * 64) return;
    int n = idx >> 6;
    int j = idx & 63;
    atomAddF(&sums[batch[n] * 64 + j], h[idx]);
}

__global__ void k_pool_cnt(const int* __restrict__ batch, float* __restrict__ cnts) {
    int n = blockIdx.x * blockDim.x + threadIdx.x;
    if (n >= N_NODES) return;
    atomAddF(&cnts[batch[n]], 1.0f);
}

// ------------------------------------------------------------------- head
__global__ void k_head(const float* __restrict__ sums, const float* __restrict__ cnts,
                       const float* __restrict__ Wf, const float* __restrict__ bf,
                       const float* __restrict__ Wl, const float* __restrict__ bl,
                       float* __restrict__ out) {
    int g = blockIdx.x * blockDim.x + threadIdx.x;
    if (g >= N_GRAPHS) return;
    float c   = cnts[g];
    float inv = 1.0f / (c > 1.0f ? c : 1.0f);
    float o   = bl[0];
#pragma unroll 1
    for (int t = 0; t < 10; ++t) {
        float a = bf[t];
        for (int k = 0; k < 64; ++k) a += sums[g * 64 + k] * inv * Wf[k * 10 + t];
        a = a > 0.0f ? a : 0.0f;
        o += a * Wl[t];
    }
    out[g] = o;
}

// ===========================================================================
extern "C" void kernel_launch(void* const* d_in, const int* in_sizes, int n_in,
                              void* d_out, int out_size, void* d_ws, size_t ws_size,
                              hipStream_t stream) {
    const float* x     = (const float*)d_in[0];
    const int*   ei    = (const int*)d_in[1];
    const int*   batch = (const int*)d_in[2];
    const float* W1    = (const float*)d_in[3];
    const float* b1    = (const float*)d_in[4];
    const float* W2    = (const float*)d_in[5];
    const float* b2    = (const float*)d_in[6];
    const float* W3    = (const float*)d_in[7];
    const float* b3    = (const float*)d_in[8];
    const float* Wf    = (const float*)d_in[9];
    const float* bf    = (const float*)d_in[10];
    const float* Wl    = (const float*)d_in[11];
    const float* bl    = (const float*)d_in[12];
    const float* eps1  = (const float*)d_in[13];
    const float* eps2  = (const float*)d_in[14];
    const float* eps3  = (const float*)d_in[15];
    float* out = (float*)d_out;

    // workspace layout (256B aligned partitions)
    char* ws = (char*)d_ws;
    float*    agg  = (float*)(ws);                               // N*128 f32
    float*    hA   = (float*)(ws + 51200000);                    // N*64  f32
    float*    hB   = (float*)(ws + 51200000 + 25600000);         // N*64  f32
    _Float16* W1t  = (_Float16*)(ws + 102400000);                // 64*128 f16
    _Float16* W2t  = (_Float16*)(ws + 102400000 + 16384);        // 64*64  f16
    _Float16* W3t  = (_Float16*)(ws + 102400000 + 16384 + 8192); // 64*64  f16
    float*    sums = (float*)(ws + 102432768);                   // G*64 f32
    float*    cnts = (float*)(ws + 102432768 + 256000);          // G    f32

    const int BLK    = 256;
    const int Mtiles = N_NODES / 16;                 // 6250
    const int gemmBlocks = (Mtiles + 7) / 8;         // 8 waves/block

    // weights -> f16 transposed
    k_wcvt<<<(64 * 128 + BLK - 1) / BLK, BLK, 0, stream>>>(W1, W1t, 128);
    k_wcvt<<<(64 * 64 + BLK - 1) / BLK, BLK, 0, stream>>>(W2, W2t, 64);
    k_wcvt<<<(64 * 64 + BLK - 1) / BLK, BLK, 0, stream>>>(W3, W3t, 64);

    // ---- layer 1 (K = 128) ----
    k_zero<<<(N_NODES * 128 + BLK - 1) / BLK, BLK, 0, stream>>>(agg, N_NODES * 128);
    {
        dim3 g((N_EDGES + BLK - 1) / BLK, 128 / 4);
        k_edge_agg<<<g, BLK, 0, stream>>>(x, ei, agg, 128);
    }
    k_gin_gemm<<<gemmBlocks, BLK, 0, stream>>>(x, agg, W1t, b1, eps1, hA, 128, Mtiles);

    // ---- layer 2 (K = 64) ----
    k_zero<<<(N_NODES * 64 + BLK - 1) / BLK, BLK, 0, stream>>>(agg, N_NODES * 64);
    {
        dim3 g((N_EDGES + BLK - 1) / BLK, 64 / 4);
        k_edge_agg<<<g, BLK, 0, stream>>>(hA, ei, agg, 64);
    }
    k_gin_gemm<<<gemmBlocks, BLK, 0, stream>>>(hA, agg, W2t, b2, eps2, hB, 64, Mtiles);

    // ---- layer 3 (K = 64) ----
    k_zero<<<(N_NODES * 64 + BLK - 1) / BLK, BLK, 0, stream>>>(agg, N_NODES * 64);
    {
        dim3 g((N_EDGES + BLK - 1) / BLK, 64 / 4);
        k_edge_agg<<<g, BLK, 0, stream>>>(hB, ei, agg, 64);
    }
    k_gin_gemm<<<gemmBlocks, BLK, 0, stream>>>(hB, agg, W3t, b3, eps3, hA, 64, Mtiles);

    // ---- pooling + head ----
    k_zero<<<(N_GRAPHS * 64 + N_GRAPHS + BLK - 1) / BLK, BLK, 0, stream>>>(sums, N_GRAPHS * 64 + N_GRAPHS);
    k_pool_sum<<<(N_NODES * 64 + BLK - 1) / BLK, BLK, 0, stream>>>(hA, batch, sums);
    k_pool_cnt<<<(N_NODES + BLK - 1) / BLK, BLK, 0, stream>>>(batch, cnts);
    k_head<<<(N_GRAPHS + BLK - 1) / BLK, BLK, 0, stream>>>(sums, cnts, Wf, bf, Wl, bl, out);
}
